// LocalLatentAttention_65274912965238
// MI455X (gfx1250) — matrix operand order repaired and checked by reading, not compiled, not run on hardware
//
#include <hip/hip_runtime.h>

// ---------------------------------------------------------------------------
// MI455X / gfx1250 implementation of LocalLatentAttention.
// All matmuls run through v_wmma_f32_16x16x32_bf16 (wave32, fp32 accumulate).
// Intermediate tensors (qkv, rq, rkv, local, latent, remote, mixed) are kept
// in HBM as bf16: every consumer is a bf16 WMMA, so this loses no additional
// precision while halving ~250MB of intermediate traffic on a 23.3 TB/s,
// bandwidth-bound workload, and it removes per-fragment f32->bf16 cvt VALU
// work from the inner loops.  GEMMs stage A/B tiles in LDS with gfx1250
// async-to-LDS copies (global_load_async_to_lds_b128 / s_wait_asynccnt).
// ---------------------------------------------------------------------------

typedef __attribute__((ext_vector_type(16))) __bf16 v16bf;
typedef __attribute__((ext_vector_type(8)))  float  v8f;

#define WMMA_BF16(a, b, c) \
  __builtin_amdgcn_wmma_f32_16x16x32_bf16(false, (a), false, (b), (short)0, (c), false, false)

#define BATCH 2
#define SEQ   4096
#define DMODEL 1024
#define NHEADS 16
#define HDIM  64
#define QKVLD 3072
#define LATD  256
#define LHEADS 4
#define NLAT  64
#define RKVLD 512

static __device__ __forceinline__ float rowmax16(float v) {
  v = fmaxf(v, __shfl_xor(v, 1, 32));
  v = fmaxf(v, __shfl_xor(v, 2, 32));
  v = fmaxf(v, __shfl_xor(v, 4, 32));
  v = fmaxf(v, __shfl_xor(v, 8, 32));
  return v;
}
static __device__ __forceinline__ float rowsum16(float v) {
  v += __shfl_xor(v, 1, 32);
  v += __shfl_xor(v, 2, 32);
  v += __shfl_xor(v, 4, 32);
  v += __shfl_xor(v, 8, 32);
  return v;
}

// gfx1250 async copy: global -> LDS, 16 bytes per lane, tracked on ASYNCcnt.
// lds = DS-space byte offset (low 32 bits of the flat pointer to __shared__).
static __device__ __forceinline__ void async_ld_b128(unsigned lds, const void* g) {
  asm volatile("global_load_async_to_lds_b128 %0, %1, off"
               :: "v"(lds), "v"(g) : "memory");
}
static __device__ __forceinline__ void wait_async0() {
  asm volatile("s_wait_asynccnt 0" ::: "memory");
}
static __device__ __forceinline__ unsigned lds_off(const void* p) {
  return (unsigned)(uintptr_t)p;
}
static __device__ __forceinline__ __bf16 to_bf16(float v)  { return (__bf16)v; }
static __device__ __forceinline__ __bf16 to_bf16(__bf16 v) { return v; }

// ---------------------------------------------------------------------------
// Generic GEMM: C[M,N] = A[M,K] @ W[K,N] + bias[N].
// TA in {float,__bf16} (A in HBM), TC in {float,__bf16} (C in HBM);
// W and bias are always f32 (model weights).
// Block = 256 threads = 8 waves; block tile 128x64; wave tile 32x32
// (2x2 WMMA frags); K step 32.  Per K-step the block async-stages
//   sA: 128x32 TA  (16KB f32 / 8KB bf16)  and  sB: 32x64 f32 (8KB)
// then feeds WMMA fragments from LDS (bf16 conversion only where TA==f32).
// Requires M%128==0, N%64==0, K%32==0 (true at all call sites).
// ---------------------------------------------------------------------------
template <typename TA, typename TC>
__global__ __launch_bounds__(256) void gemm_bias_kernel(
    const TA* __restrict__ A, const float* __restrict__ W,
    const float* __restrict__ bias, TC* __restrict__ C,
    int M, int N, int K) {
  __shared__ TA    sA[128 * 32];
  __shared__ float sB[32 * 64];

  const int tid  = threadIdx.x;
  const int lane = tid & 31;
  const int wid  = tid >> 5;
  const int wm = wid & 3;
  const int wn = wid >> 2;
  const int m0 = blockIdx.x * 128 + wm * 32;
  const int n0 = blockIdx.y * 64 + wn * 32;

  // staging: A: thread -> (row = tid/2, 16 elements); B: (row = tid/8, 8 f32)
  const int arow_s = tid >> 1;
  const int acol_s = (tid & 1) * 16;
  const int brow_s = tid >> 3;
  const int bcol_s = (tid & 7) * 8;
  const unsigned ldsA = lds_off(&sA[arow_s * 32 + acol_s]);
  const unsigned ldsB = lds_off(&sB[brow_s * 64 + bcol_s]);
  const TA* gA0 = A + (size_t)(blockIdx.x * 128 + arow_s) * K + acol_s;
  const size_t gBcol = (size_t)blockIdx.y * 64 + bcol_s;

  const int akoff = (lane >> 4) << 3;   // +0 / +8 K split (A frag layout)
  const int bkoff = (lane >> 4) << 3;   // +0 / +8 K split (B frag layout)
  const int arow_f = wm * 32 + (lane & 15);
  const int bcol_f = wn * 32 + (lane & 15);

  v8f acc00 = {}, acc01 = {}, acc10 = {}, acc11 = {};

  for (int k0 = 0; k0 < K; k0 += 32) {
    // ---- async stage A (128x32 TA) and B (32x64 f32) tiles into LDS
    const TA* gA = gA0 + k0;
    const float* gB = W + (size_t)(k0 + brow_s) * N + gBcol;
    if constexpr (sizeof(TA) == 4) {            // 16 f32 = 64B -> 4 x b128
      async_ld_b128(ldsA,      gA);
      async_ld_b128(ldsA + 16, gA + 4);
      async_ld_b128(ldsA + 32, gA + 8);
      async_ld_b128(ldsA + 48, gA + 12);
    } else {                                    // 16 bf16 = 32B -> 2 x b128
      async_ld_b128(ldsA,      gA);
      async_ld_b128(ldsA + 16, gA + 8);
    }
    async_ld_b128(ldsB,      gB);
    async_ld_b128(ldsB + 16, gB + 4);
    wait_async0();
    __syncthreads();

    // ---- fragments from LDS
    v16bf a0, a1, b0, b1;
#pragma unroll
    for (int e = 0; e < 8; ++e) {
      a0[e]     = to_bf16(sA[arow_f * 32 + akoff + e]);
      a0[e + 8] = to_bf16(sA[arow_f * 32 + akoff + 16 + e]);
      a1[e]     = to_bf16(sA[(arow_f + 16) * 32 + akoff + e]);
      a1[e + 8] = to_bf16(sA[(arow_f + 16) * 32 + akoff + 16 + e]);
    }
#pragma unroll
    for (int e = 0; e < 16; ++e) {
      b0[e] = (__bf16)sB[(bkoff + (e & 7) + ((e >> 3) << 4)) * 64 + bcol_f];
      b1[e] = (__bf16)sB[(bkoff + (e & 7) + ((e >> 3) << 4)) * 64 + bcol_f + 16];
    }
    acc00 = WMMA_BF16(a0, b0, acc00);
    acc01 = WMMA_BF16(a0, b1, acc01);
    acc10 = WMMA_BF16(a1, b0, acc10);
    acc11 = WMMA_BF16(a1, b1, acc11);
    __syncthreads();   // protect LDS tiles before next-iteration overwrite
  }

  const int rowoff = (lane >> 4) << 3;
  const int ccol   = lane & 15;
  const float bia0 = bias[n0 + ccol];
  const float bia1 = bias[n0 + 16 + ccol];
#pragma unroll
  for (int r = 0; r < 8; ++r) {
    const int mr0 = m0 + r + rowoff;
    const int mr1 = mr0 + 16;
    C[(size_t)mr0 * N + n0 + ccol]      = (TC)(acc00[r] + bia0);
    C[(size_t)mr0 * N + n0 + 16 + ccol] = (TC)(acc01[r] + bia1);
    C[(size_t)mr1 * N + n0 + ccol]      = (TC)(acc10[r] + bia0);
    C[(size_t)mr1 * N + n0 + 16 + ccol] = (TC)(acc11[r] + bia1);
  }
}

// ---------------------------------------------------------------------------
// Chunk-mean pooling: pooled[b,l,d] = mean_{c<64} x[b, l*64+c, d]  (f32)
// ---------------------------------------------------------------------------
__global__ __launch_bounds__(256) void pool_mean_kernel(
    const float* __restrict__ x, float* __restrict__ pooled) {
  int idx = blockIdx.x * blockDim.x + threadIdx.x;     // BATCH*NLAT*DMODEL
  if (idx >= BATCH * NLAT * DMODEL) return;
  int d = idx & (DMODEL - 1);
  int l = (idx >> 10) & (NLAT - 1);
  int b = idx >> 16;
  const float* p = x + ((size_t)b * SEQ + (size_t)l * 64) * DMODEL + d;
  float s = 0.f;
#pragma unroll 8
  for (int c = 0; c < 64; ++c) s += p[(size_t)c * DMODEL];
  pooled[idx] = s * (1.0f / 64.0f);
}

// ---------------------------------------------------------------------------
// Local banded attention (window=128, strict causal), qkv in bf16.
// One wave owns a 16-query tile; 5 iterations of 32 keys cover the band
// [q0-128, q0+31].  QK^T and P·V run on WMMA; P moves C-layout -> A-layout
// through a private per-wave LDS buffer (in-wave DS ordering, no barrier).
// Flash (running max/sum) softmax; rows with no allowed keys output 0.
// ---------------------------------------------------------------------------
__global__ __launch_bounds__(256) void local_attn_kernel(
    const __bf16* __restrict__ qkv, __bf16* __restrict__ outl) {
  __shared__ __bf16 sP[8][16 * 32];
  const int lane = threadIdx.x & 31;
  const int wid  = threadIdx.x >> 5;
  const int tilesPerBH = SEQ / 128;                 // 32
  const int bh  = blockIdx.x / tilesPerBH;
  const int blk = blockIdx.x % tilesPerBH;
  const int b = bh >> 4, h = bh & 15;
  const int q0 = blk * 128 + wid * 16;
  const size_t rowbase = (size_t)b * SEQ;

  // Q A-frags (16x64 -> two K=32 fragments); direct bf16 copies, no cvt
  v16bf aq[2];
  {
    const int tok = q0 + (lane & 15);
    const __bf16* pq = qkv + (rowbase + tok) * QKVLD + h * HDIM;
    const int kof = (lane >> 4) << 3;
#pragma unroll
    for (int ks = 0; ks < 2; ++ks)
#pragma unroll
      for (int e = 0; e < 8; ++e) {
        aq[ks][e]     = pq[ks * 32 + kof + e];
        aq[ks][e + 8] = pq[ks * 32 + kof + 16 + e];
      }
  }

  float mrow[8], lrow[8];
#pragma unroll
  for (int r = 0; r < 8; ++r) { mrow[r] = -3.0e38f; lrow[r] = 0.f; }
  v8f o0 = {}, o1 = {}, o2 = {}, o3 = {};
  __bf16* myP = &sP[wid][0];
  const int rowoff = (lane >> 4) << 3;

  for (int it = 0; it < 5; ++it) {
    const int kb = q0 - 128 + it * 32;
    // ---- scores: two 16x16 tiles over 32 keys
    v8f s0 = {}, s1 = {};
#pragma unroll
    for (int ks = 0; ks < 2; ++ks) {
      v16bf bk0, bk1;
      const int dof  = ks * 32 + ((lane >> 4) << 3);
      const int key0 = kb + (lane & 15);
      const int kc0  = min(max(key0, 0), SEQ - 1);
      const int kc1  = min(max(key0 + 16, 0), SEQ - 1);
      const __bf16* pk0 = qkv + (rowbase + kc0) * QKVLD + DMODEL + h * HDIM + dof;
      const __bf16* pk1 = qkv + (rowbase + kc1) * QKVLD + DMODEL + h * HDIM + dof;
#pragma unroll
      for (int e = 0; e < 16; ++e) {
        bk0[e] = pk0[(e & 7) + ((e >> 3) << 4)];
        bk1[e] = pk1[(e & 7) + ((e >> 3) << 4)];
      }
      s0 = WMMA_BF16(aq[ks], bk0, s0);
      s1 = WMMA_BF16(aq[ks], bk1, s1);
    }
    // ---- mask + flash softmax update
    float p0[8], p1[8];
    const int key0 = kb + (lane & 15);
    const int key1 = key0 + 16;
#pragma unroll
    for (int r = 0; r < 8; ++r) {
      const int q = q0 + r + rowoff;
      float v0 = s0[r] * 0.125f;                    // 1/sqrt(64)
      float v1 = s1[r] * 0.125f;
      const bool a0 = (key0 >= 0) && (key0 < q) && (q - key0 <= 128);
      const bool a1 = (key1 >= 0) && (key1 < q) && (q - key1 <= 128);
      v0 = a0 ? v0 : -3.0e38f;
      v1 = a1 ? v1 : -3.0e38f;
      const float mt   = rowmax16(fmaxf(v0, v1));
      const float mnew = fmaxf(mrow[r], mt);
      const float alpha = __expf(mrow[r] - mnew);
      const float e0 = a0 ? __expf(v0 - mnew) : 0.f;
      const float e1 = a1 ? __expf(v1 - mnew) : 0.f;
      const float rs = rowsum16(e0 + e1);
      lrow[r] = lrow[r] * alpha + rs;
      mrow[r] = mnew;
      o0[r] *= alpha; o1[r] *= alpha; o2[r] *= alpha; o3[r] *= alpha;
      p0[r] = e0; p1[r] = e1;
    }
    // ---- P: C-layout -> LDS row-major -> A-layout fragment
#pragma unroll
    for (int r = 0; r < 8; ++r) {
      const int row = r + rowoff;
      myP[row * 32 + (lane & 15)]      = (__bf16)p0[r];
      myP[row * 32 + 16 + (lane & 15)] = (__bf16)p1[r];
    }
    v16bf pa;
    {
      const int row = lane & 15;
      const int kof = (lane >> 4) << 3;
#pragma unroll
      for (int e = 0; e < 8; ++e) {
        pa[e]     = myP[row * 32 + kof + e];
        pa[e + 8] = myP[row * 32 + kof + 16 + e];
      }
    }
    // ---- O += P·V
#pragma unroll
    for (int dt = 0; dt < 4; ++dt) {
      v16bf bv;
      const int d    = dt * 16 + (lane & 15);
      const int keyb = kb + ((lane >> 4) << 3);
#pragma unroll
      for (int e = 0; e < 16; ++e) {
        const int kc = min(max(keyb + (e & 7) + ((e >> 3) << 4), 0), SEQ - 1);
        bv[e] = qkv[(rowbase + kc) * QKVLD + 2 * DMODEL + h * HDIM + d];
      }
      if (dt == 0)      o0 = WMMA_BF16(pa, bv, o0);
      else if (dt == 1) o1 = WMMA_BF16(pa, bv, o1);
      else if (dt == 2) o2 = WMMA_BF16(pa, bv, o2);
      else              o3 = WMMA_BF16(pa, bv, o3);
    }
  }

#pragma unroll
  for (int r = 0; r < 8; ++r) {
    const int q = q0 + r + rowoff;
    const float inv = lrow[r] > 0.f ? 1.0f / lrow[r] : 0.f;
    const size_t base = ((size_t)(b * NHEADS + h) * SEQ + q) * HDIM;
    outl[base + 0  + (lane & 15)] = (__bf16)(o0[r] * inv);
    outl[base + 16 + (lane & 15)] = (__bf16)(o1[r] * inv);
    outl[base + 32 + (lane & 15)] = (__bf16)(o2[r] * inv);
    outl[base + 48 + (lane & 15)] = (__bf16)(o3[r] * inv);
  }
}

// ---------------------------------------------------------------------------
// Latent attention: rq (B,S,4,64) vs 64 latents (all bf16), chunk-causal
// mask (latent j allowed iff j < s/64 — tile-uniform since 16 | 64).
// Single-pass softmax (all 64 scores in registers), then P·rv via WMMA.
// ---------------------------------------------------------------------------
__global__ __launch_bounds__(256) void latent_attn_kernel(
    const __bf16* __restrict__ rq, const __bf16* __restrict__ rkv,
    __bf16* __restrict__ lato) {
  __shared__ __bf16 sP[8][16 * 64];
  const int lane = threadIdx.x & 31;
  const int wid  = threadIdx.x >> 5;
  const int tilesPerBH = SEQ / 128;                 // 32
  const int bh  = blockIdx.x / tilesPerBH;
  const int blk = blockIdx.x % tilesPerBH;
  const int b = bh >> 2, lh = bh & 3;
  const int q0  = blk * 128 + wid * 16;
  const int cid = q0 >> 6;                          // same chunk for whole tile
  const int latidx = lane & 15;
  const int rowoff = (lane >> 4) << 3;
  const int kof    = (lane >> 4) << 3;

  // rq A-frags
  v16bf aq[2];
  {
    const int tok = q0 + latidx;
    const __bf16* pq = rq + ((size_t)b * SEQ + tok) * LATD + lh * 64;
#pragma unroll
    for (int ks = 0; ks < 2; ++ks)
#pragma unroll
      for (int e = 0; e < 8; ++e) {
        aq[ks][e]     = pq[ks * 32 + kof + e];
        aq[ks][e + 8] = pq[ks * 32 + kof + 16 + e];
      }
  }

  // scores over 4 latent tiles
  v8f sfr[4];
#pragma unroll
  for (int j = 0; j < 4; ++j) {
    v8f sj = {};
#pragma unroll
    for (int ks = 0; ks < 2; ++ks) {
      v16bf bk;
      const int lat = j * 16 + latidx;
      const int dof = ks * 32 + ((lane >> 4) << 3);
      const __bf16* pk = rkv + ((size_t)b * NLAT + lat) * RKVLD + lh * 64 + dof;
#pragma unroll
      for (int e = 0; e < 16; ++e) bk[e] = pk[(e & 7) + ((e >> 3) << 4)];
      sj = WMMA_BF16(aq[ks], bk, sj);
    }
    sfr[j] = sj;
  }

  // mask + softmax (normalized probabilities)
  float sv[4][8];
#pragma unroll
  for (int j = 0; j < 4; ++j)
#pragma unroll
    for (int r = 0; r < 8; ++r) {
      const float v = sfr[j][r] * 0.125f;           // 1/sqrt(64)
      sv[j][r] = (j * 16 + latidx < cid) ? v : -3.0e38f;
    }
#pragma unroll
  for (int r = 0; r < 8; ++r) {
    float mt = fmaxf(fmaxf(sv[0][r], sv[1][r]), fmaxf(sv[2][r], sv[3][r]));
    mt = rowmax16(mt);
    float ssum = 0.f;
#pragma unroll
    for (int j = 0; j < 4; ++j) {
      const float e = sv[j][r] > -1.0e30f ? __expf(sv[j][r] - mt) : 0.f;
      sv[j][r] = e;
      ssum += e;
    }
    ssum = rowsum16(ssum);
    const float inv = ssum > 0.f ? 1.0f / ssum : 0.f;
#pragma unroll
    for (int j = 0; j < 4; ++j) sv[j][r] *= inv;
  }

  // P -> LDS -> A-frags (K=64 in two K=32 fragments)
  __bf16* myP = &sP[wid][0];
#pragma unroll
  for (int j = 0; j < 4; ++j)
#pragma unroll
    for (int r = 0; r < 8; ++r)
      myP[(r + rowoff) * 64 + j * 16 + latidx] = (__bf16)sv[j][r];

  v16bf pa[2];
  {
    const int row = latidx;
#pragma unroll
    for (int ks = 0; ks < 2; ++ks)
#pragma unroll
      for (int e = 0; e < 8; ++e) {
        pa[ks][e]     = myP[row * 64 + ks * 32 + kof + e];
        pa[ks][e + 8] = myP[row * 64 + ks * 32 + kof + 16 + e];
      }
  }

  // out = P · rv, store directly to (B,S,LH,64) == [B*S, 256] head-major
#pragma unroll
  for (int dt = 0; dt < 4; ++dt) {
    v8f acc = {};
#pragma unroll
    for (int ks = 0; ks < 2; ++ks) {
      v16bf bv;
      const int d    = dt * 16 + latidx;
      const int latb = ks * 32 + ((lane >> 4) << 3);
#pragma unroll
      for (int e = 0; e < 16; ++e)
        bv[e] = rkv[((size_t)b * NLAT + latb + (e & 7) + ((e >> 3) << 4)) * RKVLD + LATD + lh * 64 + d];
      acc = WMMA_BF16(pa[ks], bv, acc);
    }
#pragma unroll
    for (int r = 0; r < 8; ++r) {
      const int q = q0 + r + rowoff;
      lato[((size_t)b * SEQ + q) * LATD + lh * 64 + dt * 16 + latidx] = (__bf16)acc[r];
    }
  }
}

// ---------------------------------------------------------------------------
// Gate + mix: per (b,h,s) scalar gate = sigmoid(local·gl_w + remote·gr_w),
// mixed written back (bf16) in (B,S,D) layout for the output GEMM.
// ---------------------------------------------------------------------------
__global__ __launch_bounds__(256) void gate_mix_kernel(
    const __bf16* __restrict__ loc, const __bf16* __restrict__ rem,
    const float* __restrict__ glw, const float* __restrict__ grw,
    __bf16* __restrict__ mixed) {
  int idx = blockIdx.x * blockDim.x + threadIdx.x;   // BATCH*NHEADS*SEQ
  if (idx >= BATCH * NHEADS * SEQ) return;
  const int s = idx & (SEQ - 1);
  const int h = (idx >> 12) & (NHEADS - 1);
  const int b = idx >> 16;
  const __bf16* lp = loc + ((size_t)(b * NHEADS + h) * SEQ + s) * HDIM;
  const __bf16* rp = rem + ((size_t)b * SEQ + s) * DMODEL + h * HDIM;
  float gl = 0.f, gr = 0.f;
#pragma unroll 8
  for (int d = 0; d < HDIM; ++d) {
    gl += (float)lp[d] * glw[d];
    gr += (float)rp[d] * grw[d];
  }
  const float g = 1.0f / (1.0f + __expf(-(gl + gr)));
  __bf16* mp = mixed + ((size_t)b * SEQ + s) * DMODEL + h * HDIM;
#pragma unroll 8
  for (int d = 0; d < HDIM; ++d)
    mp[d] = (__bf16)(g * (float)lp[d] + (1.0f - g) * (float)rp[d]);
}

// ---------------------------------------------------------------------------
extern "C" void kernel_launch(void* const* d_in, const int* in_sizes, int n_in,
                              void* d_out, int out_size, void* d_ws, size_t ws_size,
                              hipStream_t stream) {
  const float* x      = (const float*)d_in[0];
  const float* qkv_w  = (const float*)d_in[1];
  const float* qkv_b  = (const float*)d_in[2];
  const float* rq_w   = (const float*)d_in[3];
  const float* rq_b   = (const float*)d_in[4];
  const float* rkv_w  = (const float*)d_in[5];
  const float* rkv_b  = (const float*)d_in[6];
  const float* rout_w = (const float*)d_in[7];
  const float* rout_b = (const float*)d_in[8];
  const float* out_w  = (const float*)d_in[9];
  const float* out_b  = (const float*)d_in[10];
  const float* gl_w   = (const float*)d_in[11];
  const float* gr_w   = (const float*)d_in[12];
  float* out = (float*)d_out;

  char* wsb = (char*)d_ws;
  size_t off = 0;
  auto carve = [&](size_t bytes) -> void* {
    void* p = (void*)(wsb + off);
    off += (bytes + 255) & ~(size_t)255;
    return p;
  };
  const size_t BS = (size_t)BATCH * SEQ;            // 8192
  __bf16* qkv    = (__bf16*)carve(BS * QKVLD * 2);                 // 50.3 MB
  __bf16* rq     = (__bf16*)carve(BS * LATD * 2);                  //  4.2 MB
  float*  pooled = (float*) carve((size_t)BATCH * NLAT * DMODEL * 4);
  __bf16* rkvb   = (__bf16*)carve((size_t)BATCH * NLAT * RKVLD * 2);
  __bf16* locb   = (__bf16*)carve((size_t)BATCH * NHEADS * SEQ * HDIM * 2);
  __bf16* latb   = (__bf16*)carve(BS * LATD * 2);
  __bf16* remote = (__bf16*)carve(BS * DMODEL * 2);
  __bf16* mixed  = (__bf16*)carve(BS * DMODEL * 2);
  (void)ws_size; (void)in_sizes; (void)n_in; (void)out_size;

  const dim3 blk(256);

  // 1) qkv = x @ qkv_w + qkv_b                     [8192 x 3072, K=1024] -> bf16
  gemm_bias_kernel<float, __bf16><<<dim3(8192 / 128, 3072 / 64), blk, 0, stream>>>(
      x, qkv_w, qkv_b, qkv, 8192, 3072, 1024);
  // 2) rq = x @ rq_w + rq_b                        [8192 x 256, K=1024] -> bf16
  gemm_bias_kernel<float, __bf16><<<dim3(8192 / 128, 256 / 64), blk, 0, stream>>>(
      x, rq_w, rq_b, rq, 8192, 256, 1024);
  // 3) pooled = chunk-mean(x)                      (f32)
  pool_mean_kernel<<<dim3((BATCH * NLAT * DMODEL + 255) / 256), blk, 0, stream>>>(x, pooled);
  // 4) rkv = pooled @ rkv_w + rkv_b                [128 x 512, K=1024] -> bf16
  gemm_bias_kernel<float, __bf16><<<dim3(1, 512 / 64), blk, 0, stream>>>(
      pooled, rkv_w, rkv_b, rkvb, 128, 512, 1024);
  // 5) local banded attention                      (bf16 in/out)
  local_attn_kernel<<<dim3(BATCH * NHEADS * (SEQ / 128)), blk, 0, stream>>>(qkv, locb);
  // 6) latent attention                            (bf16 in/out)
  latent_attn_kernel<<<dim3(BATCH * LHEADS * (SEQ / 128)), blk, 0, stream>>>(rq, rkvb, latb);
  // 7) remote = lat_out @ rout_w + rout_b          [8192 x 1024, K=256] bf16 -> bf16
  gemm_bias_kernel<__bf16, __bf16><<<dim3(8192 / 128, 1024 / 64), blk, 0, stream>>>(
      latb, rout_w, rout_b, remote, 8192, 1024, 256);
  // 8) gate + mix                                  (bf16 -> bf16)
  gate_mix_kernel<<<dim3((BATCH * NHEADS * SEQ + 255) / 256), blk, 0, stream>>>(
      locb, remote, gl_w, gr_w, mixed);
  // 9) out = mixed @ out_w + out_b                 [8192 x 1024, K=1024] bf16 -> f32
  gemm_bias_kernel<__bf16, float><<<dim3(8192 / 128, 1024 / 64), blk, 0, stream>>>(
      mixed, out_w, out_b, out, 8192, 1024, 1024);
}
